// DECOLLENetwork_43671227466541
// MI455X (gfx1250) — matrix operand order repaired and checked by reading, not compiled, not run on hardware
//
#include <hip/hip_runtime.h>

// DECOLLE SNN forward for MI455X (gfx1250), wave32 + WMMA, round 2.
//
// Shapes: B=32, F_IN=512, T=256, H=1024, OUT=128 (fp32 in/out).
// d_out: s1[B*H*T] | s2[B*H*T] | r1[B*O*T] | r2[B*O*T] | v1[B*H*T] | v2[B*H*T] | c1 | c2
//
// Round-2 changes vs round 1:
//  - All GEMM operands pre-converted to f16; activations kept TRANSPOSED
//    [B, T, K] so B-fragments are contiguous 32B vector loads per lane.
//  - 2x4 register tiling per wave: 8 v_wmma per K-step, wide b128 loads,
//    zero cvt instructions in the hot loop.
//  - LIF scan writes f16 spikes straight into the transposed layout.

typedef _Float16 half16 __attribute__((ext_vector_type(16)));
typedef _Float16 half8  __attribute__((ext_vector_type(8)));
typedef float    float8 __attribute__((ext_vector_type(8)));

#define LIF_THRESH 1.25f
#define CUR_KEEP   0.75f   // 1 - CUR_DECAY
#define VOLT_KEEP  0.97f   // 1 - VOLT_DECAY

#define MT 2               // 16-row tiles per wave
#define NT 4               // 16-col tiles per wave
#define WAVES 4            // waves per block

// ---------------------------------------------------------------------------
// WMMA GEMM:  Out[b] = A (MxK f16 row-major) * Xt[b]^T, Xt stored [B, T, K].
// One wave computes a 32x64 tile via 2x4 register tiling, K stepped by 32.
// Fragment layouts used here (contiguous k order, not the ISA table order; A and B match):
//   A lane (M=lane&15): halves 0..15 = K[16h .. 16h+15].
//   B lane (N=lane&15): halves 0..15 = K[16h .. 16h+15]  (contiguous in Xt).
//   C VGPR r -> M = r + 8h, N = lane&15.
// ---------------------------------------------------------------------------
__global__ __launch_bounds__(32 * WAVES)
void gemm_wmma_f16t_kernel(const _Float16* __restrict__ A,   // [M, K]
                           const _Float16* __restrict__ Xt,  // [B, T, K]
                           float* __restrict__ Out,          // [B, M, T]
                           int M, int K, int T)
{
    const int lane = threadIdx.x & 31;
    const int m16  = lane & 15;
    const int hl   = lane >> 4;
    const int t0   = blockIdx.x * (16 * NT);
    const int m0   = (blockIdx.y * WAVES + threadIdx.y) * (16 * MT);
    const int b    = blockIdx.z;

    const _Float16* __restrict__ Xb = Xt + (size_t)b * T * K;
    float* __restrict__ Ob          = Out + (size_t)b * M * T;

    const _Float16* arow[MT];
    #pragma unroll
    for (int mt = 0; mt < MT; ++mt)
        arow[mt] = A + (size_t)(m0 + mt * 16 + m16) * K;
    const _Float16* brow[NT];
    #pragma unroll
    for (int nt = 0; nt < NT; ++nt)
        brow[nt] = Xb + (size_t)(t0 + nt * 16 + m16) * K;

    float8 acc[MT][NT] = {};

    for (int k0 = 0; k0 < K; k0 += 32) {
        half16 af[MT];
        half16 bf[NT];
        #pragma unroll
        for (int mt = 0; mt < MT; ++mt) {
            const half8 lo = *(const half8*)(arow[mt] + k0 + 16 * hl);      // 16B
            const half8 hi = *(const half8*)(arow[mt] + k0 + 16 * hl + 8);  // 16B
            af[mt] = __builtin_shufflevector(lo, hi,
                        0, 1, 2, 3, 4, 5, 6, 7, 8, 9, 10, 11, 12, 13, 14, 15);
        }
        #pragma unroll
        for (int nt = 0; nt < NT; ++nt)
            bf[nt] = *(const half16*)(brow[nt] + k0 + 16 * hl);             // 32B

        #pragma unroll
        for (int mt = 0; mt < MT; ++mt)
            #pragma unroll
            for (int nt = 0; nt < NT; ++nt)
                acc[mt][nt] = __builtin_amdgcn_wmma_f32_16x16x32_f16(
                    false, af[mt], false, bf[nt],
                    (short)0, acc[mt][nt], false, false);
    }

    #pragma unroll
    for (int mt = 0; mt < MT; ++mt) {
        #pragma unroll
        for (int r = 0; r < 8; ++r) {
            float* __restrict__ orow = Ob + (size_t)(m0 + mt * 16 + r + 8 * hl) * T;
            #pragma unroll
            for (int nt = 0; nt < NT; ++nt)
                orow[t0 + nt * 16 + m16] = acc[mt][nt][r];
        }
    }
}

// ---------------------------------------------------------------------------
// CUBA LIF recurrence, in place: buffer holds z on entry, pre-reset voltage
// on exit. Also emits f16 spikes into transposed [B, T, N] layout (writes
// for a fixed t are lane-consecutive in n -> coalesced).
// ---------------------------------------------------------------------------
__global__ void lif_scan_kernel(float* __restrict__ zv,        // [B*N, T] z -> v
                                float* __restrict__ sp,        // [B*N, T] spikes f32
                                _Float16* __restrict__ spt,    // [B, T, N] spikes f16
                                unsigned int* __restrict__ cnt,
                                int N, int T, int rows)
{
    const int idx = blockIdx.x * blockDim.x + threadIdx.x;
    if (idx >= rows) return;
    const int b = idx / N;
    const int n = idx - b * N;

    float* __restrict__ row_v = zv + (size_t)idx * T;
    float* __restrict__ row_s = sp + (size_t)idx * T;
    _Float16* __restrict__ st = spt + (size_t)b * T * N + n;

    float cur = 0.0f, volt = 0.0f;
    unsigned int c = 0;
    for (int t = 0; t < T; ++t) {
        cur = CUR_KEEP * cur + row_v[t];
        const float vpre = VOLT_KEEP * volt + cur;
        row_v[t] = vpre;                          // pre-reset voltage output
        const bool fired = (vpre >= LIF_THRESH);
        row_s[t] = fired ? 1.0f : 0.0f;
        st[(size_t)t * N] = fired ? (_Float16)1.0f : (_Float16)0.0f;
        c += fired ? 1u : 0u;
        volt = fired ? 0.0f : vpre;               // hard reset carry
    }
    atomicAdd(cnt, c);
}

// ---------------------------------------------------------------------------
// f32 -> f16 convert (weights, layout preserved)
// ---------------------------------------------------------------------------
__global__ void cvt_f32_f16_kernel(const float* __restrict__ src,
                                   _Float16* __restrict__ dst, int n)
{
    const int i = blockIdx.x * blockDim.x + threadIdx.x;
    if (i < n) dst[i] = (_Float16)src[i];
}

// ---------------------------------------------------------------------------
// [B, F, T] f32  ->  [B, T, F] f16, LDS-tiled 32x32 transpose
// ---------------------------------------------------------------------------
__global__ __launch_bounds__(256)
void transpose_cvt_kernel(const float* __restrict__ src,  // [B, F, T]
                          _Float16* __restrict__ dst,     // [B, T, F]
                          int F, int T)
{
    __shared__ _Float16 tile[32][34];   // pad -> odd dword stride, no bank conflicts
    const int b  = blockIdx.z;
    const int f0 = blockIdx.y * 32;
    const int t0 = blockIdx.x * 32;
    const int tx = threadIdx.x;         // 0..31
    const int ty = threadIdx.y;         // 0..7

    const float* __restrict__ s = src + (size_t)b * F * T;
    _Float16* __restrict__ d    = dst + (size_t)b * T * F;

    #pragma unroll
    for (int j = 0; j < 32; j += 8)
        tile[ty + j][tx] = (_Float16)s[(size_t)(f0 + ty + j) * T + t0 + tx];
    __syncthreads();
    #pragma unroll
    for (int j = 0; j < 32; j += 8)
        d[(size_t)(t0 + ty + j) * F + f0 + tx] = tile[tx][ty + j];
}

__global__ void zero_counts_kernel(unsigned int* cnts)
{
    if (threadIdx.x < 2) cnts[threadIdx.x] = 0u;
}

__global__ void finalize_counts_kernel(const unsigned int* __restrict__ cnts,
                                       float* __restrict__ out, float inv_n)
{
    if (threadIdx.x < 2) out[threadIdx.x] = (float)cnts[threadIdx.x] * inv_n;
}

// ---------------------------------------------------------------------------
extern "C" void kernel_launch(void* const* d_in, const int* in_sizes, int n_in,
                              void* d_out, int out_size, void* d_ws, size_t ws_size,
                              hipStream_t stream)
{
    (void)in_sizes; (void)n_in; (void)out_size; (void)ws_size;

    const float* spike = (const float*)d_in[0];  // [B, F, T]
    const float* W1    = (const float*)d_in[1];  // [H, F]
    const float* W2    = (const float*)d_in[2];  // [H, H]
    const float* R1    = (const float*)d_in[3];  // [O, H]
    const float* R2    = (const float*)d_in[4];  // [O, H]

    const int B = 32, F = 512, T = 256, H = 1024, O = 128;
    const size_t S   = (size_t)B * H * T;   // 8,388,608
    const size_t RSZ = (size_t)B * O * T;   // 1,048,576

    float* out = (float*)d_out;
    float* s1  = out;
    float* s2  = out + S;
    float* r1  = out + 2 * S;
    float* r2  = out + 2 * S + RSZ;
    float* v1  = out + 2 * S + 2 * RSZ;
    float* v2  = v1 + S;
    float* cts = v1 + 2 * S;                // 2 floats

    // Workspace carve-up (all 32B aligned; ~46 MB total).
    unsigned int* icnt = (unsigned int*)d_ws;       // 2 uints in first 32B
    _Float16* wsh = (_Float16*)d_ws;
    size_t off = 16;                                // skip 32B count region
    _Float16* W1h = wsh + off; off += (size_t)H * F;
    _Float16* W2h = wsh + off; off += (size_t)H * H;
    _Float16* R1h = wsh + off; off += (size_t)O * H;
    _Float16* R2h = wsh + off; off += (size_t)O * H;
    _Float16* Xh  = wsh + off; off += (size_t)B * T * F;   // spike^T f16
    _Float16* S1h = wsh + off; off += (size_t)B * T * H;   // s1^T f16
    _Float16* S2h = wsh + off; off += (size_t)B * T * H;   // s2^T f16

    zero_counts_kernel<<<1, 32, 0, stream>>>(icnt);

    // Pre-convert weights to f16 (one-time, ~3.5 MB).
    cvt_f32_f16_kernel<<<(H * F + 255) / 256, 256, 0, stream>>>(W1, W1h, H * F);
    cvt_f32_f16_kernel<<<(H * H + 255) / 256, 256, 0, stream>>>(W2, W2h, H * H);
    cvt_f32_f16_kernel<<<(O * H + 255) / 256, 256, 0, stream>>>(R1, R1h, O * H);
    cvt_f32_f16_kernel<<<(O * H + 255) / 256, 256, 0, stream>>>(R2, R2h, O * H);

    // spike [B,F,T] f32 -> Xh [B,T,F] f16.
    transpose_cvt_kernel<<<dim3(T / 32, F / 32, B), dim3(32, 8), 0, stream>>>(
        spike, Xh, F, T);

    const dim3 gblk(32, WAVES, 1);
    const int TN = 16 * NT;        // 64 cols per wave
    const int TM = 16 * MT * WAVES; // 128 rows per block

    // Layer 1: z1 = W1 * spike  (into v1 slot)
    gemm_wmma_f16t_kernel<<<dim3(T / TN, H / TM, B), gblk, 0, stream>>>(
        W1h, Xh, v1, H, F, T);
    lif_scan_kernel<<<(B * H + 255) / 256, 256, 0, stream>>>(
        v1, s1, S1h, icnt + 0, H, T, B * H);

    // Layer 2: z2 = W2 * s1  (into v2 slot)
    gemm_wmma_f16t_kernel<<<dim3(T / TN, H / TM, B), gblk, 0, stream>>>(
        W2h, S1h, v2, H, H, T);
    lif_scan_kernel<<<(B * H + 255) / 256, 256, 0, stream>>>(
        v2, s2, S2h, icnt + 1, H, T, B * H);

    // Readouts: r1 = R1 * s1, r2 = R2 * s2   (O/TM = 1 block row)
    gemm_wmma_f16t_kernel<<<dim3(T / TN, O / TM, B), gblk, 0, stream>>>(
        R1h, S1h, r1, O, H, T);
    gemm_wmma_f16t_kernel<<<dim3(T / TN, O / TM, B), gblk, 0, stream>>>(
        R2h, S2h, r2, O, H, T);

    // Counts: mean spike rate = total / (B*H*T)
    finalize_counts_kernel<<<1, 32, 0, stream>>>(icnt, cts, 1.0f / (float)S);
}
